// CoPELlamaAttention_68307159875895
// MI455X (gfx1250) — matrix operand order repaired and checked by reading, not compiled, not run on hardware
//
#include <hip/hip_runtime.h>
#include <hip/hip_bf16.h>

// ---------------------------------------------------------------------------
// CoPE Llama attention for gfx1250 (MI455X):
//   wave32 + v_wmma_f32_16x16x32_f16 for all matrix math,
//   global_load_async_to_lds_b128 / s_wait_asynccnt for operand staging,
//   loops peeled so the steady state is branch-free.
// B=2, S=1024, HID=2048, H=16, KVH=8, D=128, NPOS=64, GROUPS=2.
// ---------------------------------------------------------------------------

typedef __attribute__((ext_vector_type(16))) _Float16 v16h;
typedef __attribute__((ext_vector_type(8)))  _Float16 v8h;
typedef __attribute__((ext_vector_type(8)))  float    v8f;

#define B_    2
#define S_    1024
#define HID_  2048
#define H_    16
#define KVH_  8
#define D_    128
#define NPOS_ 64
#define SCALE_ 0.08838834764831845f   // 1/sqrt(128)

#define ZERO8 {0.f,0.f,0.f,0.f,0.f,0.f,0.f,0.f}

// ---------------- CDNA5 async copy: global -> LDS (16 bytes) ---------------
// ISA 10: GLOBAL_LOAD_ASYNC_TO_LDS_B128, tracked with ASYNCcnt.
// LDS address = low 32 bits of the flat address.
__device__ __forceinline__ void async_g2l_16B(uint32_t lds_addr, const void* gaddr) {
    asm volatile("global_load_async_to_lds_b128 %0, %1, off"
                 :: "v"(lds_addr), "v"((unsigned long long)(size_t)gaddr)
                 : "memory");
}
#define ASYNC_WAIT_0() asm volatile("s_wait_asynccnt 0" ::: "memory")
#define ASYNC_WAIT_1() asm volatile("s_wait_asynccnt 1" ::: "memory")
#define ASYNC_WAIT_2() asm volatile("s_wait_asynccnt 2" ::: "memory")

__device__ __forceinline__ uint32_t lds_addr_of(const void* p) {
    return (uint32_t)(size_t)p;   // low 32 bits of flat addr == LDS byte offset
}

// ---------------- fragment loaders (contiguous k order, same for A and B) --------
// A (16x32 f16, MxK): lane L: M = L&15, element e: K = e + 16*(L>>4)
//  -> 32 contiguous bytes per lane (2 x v8h).
__device__ __forceinline__ v16h frag_a_rowmajor(const _Float16* src, int ld, int kBase) {
    int lane = threadIdx.x & 31;
    int m  = lane & 15;
    int hi = lane >> 4;
    const _Float16* p = src + m * ld + kBase + 16 * hi;
    v8h lo = *(const v8h*)(p);        // e = 0..7
    v8h hq = *(const v8h*)(p + 8);    // e = 8..15
    v16h a;
#pragma unroll
    for (int i = 0; i < 8; ++i) { a[i] = lo[i]; a[i + 8] = hq[i]; }
    return a;
}

// B (32x16 f16, KxN): lane L: N = L&15, element e: K = e + 16*(L>>4).
// Source row-major [N][Kdim] -> 32 contiguous bytes per lane (2 x v8h, 16B aligned).
__device__ __forceinline__ v16h frag_b_from_rows(const _Float16* src, int ld, int nBase, int kBase) {
    int lane = threadIdx.x & 31;
    const _Float16* p = src + (size_t)(nBase + (lane & 15)) * ld + kBase + 16 * (lane >> 4);
    v8h lo = *(const v8h*)(p);
    v8h hq = *(const v8h*)(p + 8);
    v16h b;
#pragma unroll
    for (int i = 0; i < 8; ++i) { b[i] = lo[i]; b[i + 8] = hq[i]; }
    return b;
}

// Source row-major [Kdim][N] (V chunk in LDS, pos_emb in LDS) -> strided gather.
__device__ __forceinline__ v16h frag_b_from_cols(const _Float16* src, int ld, int kBase, int nBase) {
    int lane = threadIdx.x & 31;
    int n  = nBase + (lane & 15);
    int k0 = kBase + 16 * (lane >> 4);
    v16h b;
#pragma unroll
    for (int e = 0; e < 16; ++e) b[e] = src[(k0 + e) * ld + n];
    return b;
}

__device__ __forceinline__ v8f wmma_f16(v16h a, v16h b, v8f c) {
    return __builtin_amdgcn_wmma_f32_16x16x32_f16(false, a, false, b, (short)0, c, false, false);
}

// ---------------- fp32 -> f16 convert --------------------------------------
__global__ __launch_bounds__(256) void cvt_f32_to_f16(const float* __restrict__ in,
                                                      _Float16* __restrict__ out, int n) {
    int i = blockIdx.x * 256 + threadIdx.x;
    if (i < n) out[i] = (_Float16)in[i];
}

// ---------------- WMMA GEMM: C[M,N] = X[M,K] * W[N,K]^T --------------------
// block = 256 (8 waves). Block tile: 128 M x 64 N; wave w owns rows w*16..+16.
// W tile (64x32) async-staged to LDS, double buffered, shared by all 8 waves
// (8x cut of B-operand global traffic). A fragments stream from global.
// K-loop peeled: steady state issues next stage + s_wait_asynccnt 1 with no
// branches; epilogue iteration waits asynccnt 0. mode 0: fp32 row-major out;
// mode 1: f16 scatter to [B, Hn, S, D].
#define WLD 40   // padded LDS row stride (halves) for the 64x32 W tile

__global__ __launch_bounds__(256) void gemm_wmma_xwT(
    const _Float16* __restrict__ X, const _Float16* __restrict__ W,
    _Float16* __restrict__ outH, float* __restrict__ outF,
    int M, int N, int Kd, int mode, int Hn, int Sdim, int Ddim) {
    extern __shared__ char gsm[];
    _Float16* Wbuf = (_Float16*)gsm;            // 2 x [64][WLD] halves

    int tid  = threadIdx.x;
    int wave = tid >> 5;
    int lane = tid & 31;
    int tileM  = blockIdx.y * 128 + wave * 16;
    int nBase0 = blockIdx.x * 64;
    const _Float16* Xt = X + (size_t)tileM * Kd;

    // staging map: 256 threads x 16B == 64 rows x 32 halves
    int srow = tid >> 2;                        // 0..63
    int sseg = tid & 3;                         // 0..3 (8 halves each)
    const _Float16* wsrc = W + (size_t)(nBase0 + srow) * Kd + sseg * 8;
    uint32_t wdst0 = lds_addr_of(Wbuf + srow * WLD + sseg * 8);
    uint32_t wdst1 = wdst0 + 64 * WLD * 2;

    async_g2l_16B(wdst0, wsrc);                 // prologue: k = 0 -> buf 0

    v8f acc[4] = {ZERO8, ZERO8, ZERO8, ZERO8};

    // steady state (branch-free): iterations 0 .. Kd/32 - 2
    int k = 0;
    for (; k + 32 < Kd; k += 32) {
        int cur = (k >> 5) & 1;
        async_g2l_16B(cur ? wdst0 : wdst1, wsrc + (k + 32));   // next tile
        __builtin_prefetch(Xt + (size_t)(lane & 15) * Kd + k + 64, 0, 1);
        v16h a = frag_a_rowmajor(Xt, Kd, k);    // A from global (no reuse)
        ASYNC_WAIT_1();
        __syncthreads();
        const _Float16* Wt = Wbuf + cur * 64 * WLD;
#pragma unroll
        for (int t = 0; t < 4; ++t) {
            v16h b = frag_b_from_rows(Wt, WLD, t * 16, 0);
            acc[t] = wmma_f16(a, b, acc[t]);
        }
        __syncthreads();
    }
    // epilogue: last K tile
    {
        int cur = (k >> 5) & 1;
        v16h a = frag_a_rowmajor(Xt, Kd, k);
        ASYNC_WAIT_0();
        __syncthreads();
        const _Float16* Wt = Wbuf + cur * 64 * WLD;
#pragma unroll
        for (int t = 0; t < 4; ++t) {
            v16h b = frag_b_from_rows(Wt, WLD, t * 16, 0);
            acc[t] = wmma_f16(a, b, acc[t]);
        }
    }

    int hi = lane >> 4, nl = lane & 15;
#pragma unroll
    for (int t = 0; t < 4; ++t) {
        int n = nBase0 + t * 16 + nl;
#pragma unroll
        for (int r = 0; r < 8; ++r) {
            int m = tileM + r + 8 * hi;
            float v = acc[t][r];
            if (mode == 0) {
                outF[(size_t)m * N + n] = v;
            } else {
                int bb = m / Sdim, ss = m - bb * Sdim;
                int hh = n / Ddim, dd = n - hh * Ddim;
                outH[(((size_t)bb * Hn + hh) * Sdim + ss) * Ddim + dd] = (_Float16)v;
            }
        }
    }
}

// ---------------- fused CoPE attention -------------------------------------
// grid = (S/16, H, B), block = 256 (8 waves).
// LDS: logits fp32 [16][1028], P f16 [16][1032], Qtile f16 [16][128],
//      logits_int fp32 [16][68], pos_emb f16 [128][64],
//      V chunk double buffer f16 2 x [32][VLD]  -> ~138 KB (< 320 KB / WGP).
#define LGLD 1028
#define PLD  1032
#define LILD 68
#define VLD  136

__global__ __launch_bounds__(256) void attn_cope_wmma(
    const _Float16* __restrict__ Qf, const _Float16* __restrict__ Kf,
    const _Float16* __restrict__ Vf, const _Float16* __restrict__ peH,
    _Float16* __restrict__ attnOut) {
    extern __shared__ char smem[];
    float*    lg  = (float*)smem;                  // [16][LGLD]
    _Float16* P   = (_Float16*)(lg + 16 * LGLD);   // [16][PLD]
    _Float16* Qt  = P + 16 * PLD;                  // [16][128]
    float*    li  = (float*)(Qt + 16 * 128);       // [16][LILD]
    _Float16* peL = (_Float16*)(li + 16 * LILD);   // [128][64]
    _Float16* Vb  = peL + 128 * NPOS_;             // 2 x [32][VLD]

    int tq = blockIdx.x, h = blockIdx.y, bb = blockIdx.z;
    int hk = h >> 1;                               // GQA: GROUPS = 2
    int tid  = threadIdx.x;
    int wave = tid >> 5, lane = tid & 31;
    int hi = lane >> 4, nl = lane & 15;
    int nchunks  = (tq * 16 + 16 + 31) >> 5;       // 32-key chunks covering k<=qmax
    int keytiles = nchunks * 2;

    const _Float16* Qbase = Qf + (((size_t)bb * H_ + h) * S_ + tq * 16) * D_;
    const _Float16* Kbase = Kf + (((size_t)bb * KVH_ + hk) * S_) * D_;
    const _Float16* Vbase = Vf + (((size_t)bb * KVH_ + hk) * S_) * D_;

    // stage Q tile (16x128) and pos_emb (128x64) into LDS
    for (int i = tid; i < 16 * D_; i += 256) Qt[i] = Qbase[i];
    for (int i = tid; i < D_ * NPOS_; i += 256) peL[i] = peH[i];
    __syncthreads();

    // ---- phase 1a: logits = SCALE * Q K^T  (waves round-robin key tiles)
    for (int kt = wave; kt < keytiles; kt += 8) {
        v8f acc = ZERO8;
#pragma unroll
        for (int dc = 0; dc < 4; ++dc) {
            v16h a = frag_a_rowmajor(Qt, D_, dc * 32);
            v16h b = frag_b_from_rows(Kbase, D_, kt * 16, dc * 32);
            acc = wmma_f16(a, b, acc);
        }
#pragma unroll
        for (int r = 0; r < 8; ++r)
            lg[(r + 8 * hi) * LGLD + kt * 16 + nl] = acc[r] * SCALE_;
    }
    // ---- phase 1b: logits_int = Q @ pos_emb  (waves 0..3, one 16-col tile each)
    if (wave < 4) {
        v8f acc = ZERO8;
#pragma unroll
        for (int dc = 0; dc < 4; ++dc) {
            v16h a = frag_a_rowmajor(Qt, D_, dc * 32);
            v16h b = frag_b_from_cols(peL, NPOS_, dc * 32, wave * 16);
            acc = wmma_f16(a, b, acc);
        }
#pragma unroll
        for (int r = 0; r < 8; ++r)
            li[(r + 8 * hi) * LILD + wave * 16 + nl] = acc[r];
    }
    __syncthreads();

    // ---- phase 2: gates -> reverse-cumsum positions -> CoPE add -> softmax
    // wave w owns rows {w, w+8}; chunks processed high-k -> low-k.
    for (int rr = 0; rr < 2; ++rr) {
        int row = wave + rr * 8;
        int qg  = tq * 16 + row;
        float run = 0.f;
        for (int c = nchunks - 1; c >= 0; --c) {
            int k = c * 32 + lane;
            float logit  = lg[row * LGLD + k];
            bool  masked = (k > qg);               // additive causal mask
            float gate   = masked ? 0.f : 1.f / (1.f + __expf(-logit));
            float x = gate;                        // wave32 inclusive prefix scan
            for (int off = 1; off < 32; off <<= 1) {
                float y = __shfl_up(x, off);
                if (lane >= off) x += y;
            }
            float total = __shfl(x, 31);
            float pos   = run + (total - x + gate); // inclusive suffix sum
            run += total;
            pos = fminf(pos, (float)(NPOS_ - 1));
            float pf = floorf(pos);
            int   ic = (int)ceilf(pos);
            int   fl = (int)pf;
            float w  = pos - pf;
            float lc = li[row * LILD + ic];
            float lf = li[row * LILD + fl];
            float nv = logit + lc * w + lf * (1.f - w);
            lg[row * LGLD + k] = masked ? -3.0e38f : nv;
        }
        float mx = -3.0e38f;
        for (int c = 0; c < nchunks; ++c)
            mx = fmaxf(mx, lg[row * LGLD + c * 32 + lane]);
        for (int off = 16; off; off >>= 1) mx = fmaxf(mx, __shfl_xor(mx, off));
        float sum = 0.f;
        for (int c = 0; c < nchunks; ++c) {
            int k = c * 32 + lane;
            float e = __expf(lg[row * LGLD + k] - mx);
            lg[row * LGLD + k] = e;
            sum += e;
        }
        for (int off = 16; off; off >>= 1) sum += __shfl_xor(sum, off);
        float inv = 1.f / sum;
        for (int c = 0; c < nchunks; ++c) {
            int k = c * 32 + lane;
            P[row * PLD + k] = (_Float16)(lg[row * LGLD + k] * inv);
        }
    }
    __syncthreads();

    // ---- phase 3: out = P @ V ; wave w owns d-tile w (16 cols of D=128).
    // V chunks (32 keys x 128 d) async-staged to LDS, double buffered; chunk
    // loop peeled so the steady state issues next stage + s_wait_asynccnt 2
    // with no branches, epilogue waits asynccnt 0.
    int vkey = tid >> 4;                           // 0..15
    int vseg = tid & 15;                           // 0..15 (8 halves each)
    uint32_t vb0 = lds_addr_of(Vb + vkey * VLD + vseg * 8);
    uint32_t vb1 = vb0 + 32 * VLD * 2;
    uint32_t vrow16 = 16 * VLD * 2;                // LDS offset of keys 16..31
    const _Float16* vsrc = Vbase + (size_t)vkey * D_ + vseg * 8;

    // prologue: chunk 0 -> buf 0 (2 async ops per thread)
    async_g2l_16B(vb0, vsrc);
    async_g2l_16B(vb0 + vrow16, vsrc + 16 * D_);

    v8f acc = ZERO8;
    int c = 0;
    for (; c + 1 < nchunks; ++c) {                 // steady state (branch-free)
        int cur = c & 1;
        uint32_t dst = cur ? vb0 : vb1;
        const _Float16* s = vsrc + (size_t)(c + 1) * 32 * D_;
        async_g2l_16B(dst, s);
        async_g2l_16B(dst + vrow16, s + 16 * D_);
        v16h a = frag_a_rowmajor(P, PLD, c * 32);  // P from LDS (stable)
        ASYNC_WAIT_2();
        __syncthreads();
        v16h b = frag_b_from_cols(Vb + cur * 32 * VLD, VLD, 0, wave * 16);
        acc = wmma_f16(a, b, acc);
        __syncthreads();
    }
    {                                              // epilogue: last chunk
        int cur = c & 1;
        v16h a = frag_a_rowmajor(P, PLD, c * 32);
        ASYNC_WAIT_0();
        __syncthreads();
        v16h b = frag_b_from_cols(Vb + cur * 32 * VLD, VLD, 0, wave * 16);
        acc = wmma_f16(a, b, acc);
    }
#pragma unroll
    for (int r = 0; r < 8; ++r) {
        int qg = tq * 16 + r + 8 * hi;
        attnOut[((size_t)bb * S_ + qg) * (H_ * D_) + h * D_ + wave * 16 + nl] = (_Float16)acc[r];
    }
}

// ---------------------------------------------------------------------------
extern "C" void kernel_launch(void* const* d_in, const int* in_sizes, int n_in,
                              void* d_out, int out_size, void* d_ws, size_t ws_size,
                              hipStream_t stream) {
    const float* hidden  = (const float*)d_in[0];
    // d_in[1] = attention_mask: exactly the HF causal mask -> applied analytically.
    const float* wq      = (const float*)d_in[2];
    const float* wk      = (const float*)d_in[3];
    const float* wv      = (const float*)d_in[4];
    const float* wo      = (const float*)d_in[5];
    const float* pos_emb = (const float*)d_in[6];

    const int MROWS = B_ * S_;                             // 2048
    const size_t nHID  = (size_t)MROWS * HID_;             // 4M
    const size_t nWq   = (size_t)(H_  * D_) * HID_;        // 4M
    const size_t nWk   = (size_t)(KVH_* D_) * HID_;        // 2M
    const size_t nPe   = (size_t)D_ * NPOS_;               // 8K
    const size_t nQ    = (size_t)B_ * H_  * S_ * D_;       // 4M
    const size_t nKV   = (size_t)B_ * KVH_* S_ * D_;       // 2M
    const size_t nAttn = (size_t)B_ * S_ * H_ * D_;        // 4M

    _Float16* p   = (_Float16*)d_ws;
    _Float16* Xh  = p;  p += nHID;
    _Float16* wqh = p;  p += nWq;
    _Float16* wkh = p;  p += nWk;
    _Float16* wvh = p;  p += nWk;
    _Float16* woh = p;  p += nWq;
    _Float16* peh = p;  p += nPe;
    _Float16* Qf  = p;  p += nQ;
    _Float16* Kf  = p;  p += nKV;
    _Float16* Vf  = p;  p += nKV;
    _Float16* aO  = p;  p += nAttn;

    auto cvt = [&](const float* src, _Float16* dst, size_t n) {
        cvt_f32_to_f16<<<dim3((unsigned)((n + 255) / 256)), dim3(256), 0, stream>>>(src, dst, (int)n);
    };
    cvt(hidden,  Xh,  nHID);
    cvt(wq,      wqh, nWq);
    cvt(wk,      wkh, nWk);
    cvt(wv,      wvh, nWk);
    cvt(wo,      woh, nWq);
    cvt(pos_emb, peh, nPe);

    size_t gemmSmem = (size_t)2 * 64 * WLD * 2;            // 10 KB double-buffered W tile

    // QKV projections (f16 WMMA, f32 accumulate), scattered to [B,Hn,S,D] f16
    gemm_wmma_xwT<<<dim3(HID_ / 64, MROWS / 128), dim3(256), gemmSmem, stream>>>(
        Xh, wqh, Qf, nullptr, MROWS, H_ * D_,   HID_, 1, H_,   S_, D_);
    gemm_wmma_xwT<<<dim3((KVH_ * D_) / 64, MROWS / 128), dim3(256), gemmSmem, stream>>>(
        Xh, wkh, Kf, nullptr, MROWS, KVH_ * D_, HID_, 1, KVH_, S_, D_);
    gemm_wmma_xwT<<<dim3((KVH_ * D_) / 64, MROWS / 128), dim3(256), gemmSmem, stream>>>(
        Xh, wvh, Vf, nullptr, MROWS, KVH_ * D_, HID_, 1, KVH_, S_, D_);

    // fused CoPE attention
    size_t smemBytes = (size_t)16 * LGLD * 4               // logits fp32
                     + (size_t)16 * PLD  * 2               // probabilities f16
                     + (size_t)16 * 128  * 2               // Q tile f16
                     + (size_t)16 * LILD * 4               // logits_int fp32
                     + (size_t)128 * NPOS_ * 2             // pos_emb f16
                     + (size_t)2 * 32 * VLD * 2;           // V chunk double buffer
    attn_cope_wmma<<<dim3(S_ / 16, H_, B_), dim3(256), smemBytes, stream>>>(
        Qf, Kf, Vf, peh, aO);

    // output projection -> fp32 d_out [B,S,HID]
    gemm_wmma_xwT<<<dim3(HID_ / 64, MROWS / 128), dim3(256), gemmSmem, stream>>>(
        aO, woh, nullptr, (float*)d_out, MROWS, HID_, H_ * D_, 0, 0, 0, 0);
}
